// Attention_27109833572787
// MI455X (gfx1250) — matrix operand order repaired and checked by reading, not compiled, not run on hardware
//
#include <hip/hip_runtime.h>
#include <math.h>

#define DEVINL __device__ __forceinline__

typedef __bf16 bf16_t;
typedef __attribute__((ext_vector_type(16))) __bf16 v16bf;
typedef __attribute__((ext_vector_type(8)))  __bf16 v8bf;
typedef __attribute__((ext_vector_type(8)))  float  v8f;

static constexpr int BB = 2;      // batch
static constexpr int LL = 2048;   // seq len
static constexpr int NHH = 32;    // heads
static constexpr int HDD = 128;   // head dim
static constexpr int CC = NHH * HDD;   // 4096
static constexpr int MM = BB * LL;     // 4096 GEMM M
static constexpr int KK = CC;          // 4096 GEMM K
static constexpr int NN = CC;          // 4096 GEMM N

DEVINL v8f vzero() {
  v8f v;
#pragma unroll
  for (int i = 0; i < 8; ++i) v[i] = 0.0f;
  return v;
}

// generic -> LDS(addrspace 3) byte offset for async-copy VDST operand
DEVINL unsigned lds_off_of(const void* p) {
  return (unsigned)(unsigned long long)(__attribute__((address_space(3))) const char*)p;
}
// ASYNCcnt path: DMA 16B global -> LDS without touching VGPdata
DEVINL void async_cp16(unsigned lds_byte_off, const void* gaddr) {
  asm volatile("global_load_async_to_lds_b128 %0, %1, off"
               :: "v"(lds_byte_off), "v"(gaddr)
               : "memory");
}
DEVINL void wait_async_le(int n) {
  asm volatile("s_wait_asynccnt %0" :: "i"(n) : "memory");
}

// ---- WMMA fragment loaders ------------------------------------------------
// A (16x32 bf16): lane l<16 holds row l with K {0..7,16..23}; lane l+16 holds
// row l with K {8..15,24..31}. -> two 16B chunks at (lh*8) and (16+lh*8).
DEVINL v16bf a_frag_f32(const float* __restrict__ row, int kb, int lh) {
  const float* p0 = row + kb + lh * 8;
  const float* p1 = row + kb + 16 + lh * 8;
  v16bf a;
#pragma unroll
  for (int i = 0; i < 8; ++i) { a[i] = (__bf16)p0[i]; a[i + 8] = (__bf16)p1[i]; }
  return a;
}
DEVINL v16bf a_frag_bf16(const bf16_t* __restrict__ row, int kb, int lh) {
  v8bf lo = *(const v8bf*)(row + kb + lh * 8);
  v8bf hi = *(const v8bf*)(row + kb + 16 + lh * 8);
  v16bf a;
#pragma unroll
  for (int i = 0; i < 8; ++i) { a[i] = lo[i]; a[i + 8] = hi[i]; }
  return a;
}
// B (32x16 bf16): lane l holds column (l&15); lanes 0-15 K {0..7,16..23},
// lanes 16-31 K {8..15,24..31} -> two 16B chunks at (lh*8) and (16+lh*8).
DEVINL v16bf b_frag_f32(const float* row, int kb, int lh) {
  const float* p0 = row + kb + lh * 8;
  const float* p1 = row + kb + 16 + lh * 8;
  v16bf b;
#pragma unroll
  for (int i = 0; i < 8; ++i) { b[i] = (__bf16)p0[i]; b[i + 8] = (__bf16)p1[i]; }
  return b;
}
DEVINL v16bf b_frag_bf16(const bf16_t* __restrict__ row, int kb, int lh) {
  v8bf lo = *(const v8bf*)(row + kb + lh * 8);
  v8bf hi = *(const v8bf*)(row + kb + 16 + lh * 8);
  v16bf b;
#pragma unroll
  for (int i = 0; i < 8; ++i) { b[i] = lo[i]; b[i + 8] = hi[i]; }
  return b;
}
DEVINL v8f wmma_bf16(v16bf a, v16bf b, v8f c) {
  return __builtin_amdgcn_wmma_f32_16x16x32_bf16(false, a, false, b, (short)0, c,
                                                 false, false);
}

// ---- GEMM: D[m,n] = sum_k A[m,k] * W[n,k]  (i.e. A @ W^T) ------------------
// Block = 8 waves (256 thr). All waves share one n-strip of 128 columns; the
// fp32 weight tile [128n x 32k] (16 KB) is staged into double-buffered LDS
// with global_load_async_to_lds_b128 (4 async instrs/wave/tile), overlapped
// with WMMA of the previous tile via s_wait_asynccnt. Each wave owns 16 M rows.
// MODE 0: out bf16 [B,H,L,HD] with fused RoPE (Q/K projections)
// MODE 2: out bf16 [B,H,HD,L] transposed (V projection)
// MODE 3: A is bf16 (attention output), out fp32 row-major [M,N] (out-proj)
template <int MODE>
__global__ __launch_bounds__(256) void gemm_wmma(const void* __restrict__ Ain,
                                                 const float* __restrict__ W,
                                                 void* __restrict__ Out,
                                                 const float* __restrict__ rope) {
  __shared__ __align__(128) float Wt[2][128 * 32];  // 2 x 16 KB
  const int tid = threadIdx.x;
  const int lane = tid & 31;
  const int wave = tid >> 5;
  const int col = lane & 15;
  const int lh = lane >> 4;
  const int m0 = blockIdx.x * 128 + wave * 16;
  const int n0 = blockIdx.y * 128;

  const float* Af = (const float*)Ain;
  const bf16_t* Ab = (const bf16_t*)Ain;
  const size_t arow = (size_t)(m0 + col) * KK;

  const unsigned ldsb0 = lds_off_of(&Wt[0][0]);
  const unsigned ldsb1 = lds_off_of(&Wt[1][0]);

  // stage tile for k-block `kb` into buffer `buf`: 1024 16B chunks, 4/thread
  auto issue_tile = [&](int kb, int buf) {
    const unsigned base = buf ? ldsb1 : ldsb0;
#pragma unroll
    for (int j = 0; j < 4; ++j) {
      const int c = tid + j * 256;     // chunk id
      const int row = c >> 3;          // 8 chunks (32 f32) per row
      const int k4 = (c & 7) * 4;
      const void* g = (const void*)(W + (size_t)(n0 + row) * KK + kb + k4);
      async_cp16(base + (unsigned)(row * 32 + k4) * 4u, g);
    }
  };

  v8f acc[8];
#pragma unroll
  for (int t = 0; t < 8; ++t) acc[t] = vzero();

  issue_tile(0, 0);

  const int NKB = KK / 32;  // 128
#pragma unroll 1
  for (int ki = 0; ki < NKB; ++ki) {
    const int kb = ki * 32;
    const int cur = ki & 1;
    if (ki + 1 < NKB) { issue_tile(kb + 32, cur ^ 1); wait_async_le(4); }
    else              { wait_async_le(0); }
    __syncthreads();  // tile `cur` fully resident for all waves

    v16bf a;
    if constexpr (MODE == 3) a = a_frag_bf16(Ab + arow, kb, lh);
    else                     a = a_frag_f32(Af + arow, kb, lh);

    const float* tile = &Wt[cur][0];
#pragma unroll
    for (int t = 0; t < 8; ++t) {
      v16bf b = b_frag_f32(tile + (t * 16 + col) * 32, 0, lh);
      acc[t] = wmma_bf16(a, b, acc[t]);
    }
    __syncthreads();  // everyone done reading before buffer is re-filled
  }

  // C/D layout: VGPR r -> row (m0 + r + 8*lh), column (n0 + t*16 + col)
#pragma unroll
  for (int t = 0; t < 8; ++t) {
    const int n = n0 + t * 16 + col;
#pragma unroll
    for (int r = 0; r < 8; ++r) {
      const int m = m0 + r + 8 * lh;
      float val = acc[t][r];
      if constexpr (MODE == 0) {
        const int h = n >> 7, d = n & 127;
        const int bi = m >> 11, l = m & (LL - 1);
        const float co = rope[((size_t)l * 64 + (d >> 1)) * 2 + 0];
        const float si = rope[((size_t)l * 64 + (d >> 1)) * 2 + 1];
        // even/odd head-dim pair lives in adjacent lanes
        const float part = __shfl_xor(val, 1, 32);
        const float res = (d & 1) ? (val * co + part * si) : (val * co - part * si);
        ((bf16_t*)Out)[(((size_t)bi * NHH + h) * LL + l) * HDD + d] = (__bf16)res;
      } else if constexpr (MODE == 2) {
        const int h = n >> 7, d = n & 127;
        const int bi = m >> 11, l = m & (LL - 1);
        ((bf16_t*)Out)[(((size_t)bi * NHH + h) * HDD + d) * LL + l] = (__bf16)val;
      } else {
        ((float*)Out)[(size_t)m * NN + n] = val;
      }
    }
  }
}

// ---- Flash attention: one wave handles 16 queries of one (b,h) -------------
// K-step = 32 keys: 8 wmma for S = Q K^T (HD=128), online softmax with
// 16-lane shuffle reductions, P transposed D->A layout through LDS,
// 8 wmma for O += P V (V pre-transposed so B-fragments are contiguous).
__global__ __launch_bounds__(32) void flash_attn_wmma(const bf16_t* __restrict__ Q,
                                                      const bf16_t* __restrict__ K,
                                                      const bf16_t* __restrict__ Vt,
                                                      bf16_t* __restrict__ Y) {
  __shared__ __align__(64) __bf16 Pl[16][32];
  const int lane = threadIdx.x & 31;
  const int col = lane & 15;
  const int lh = lane >> 4;
  const int nqb = LL / 16;
  int idx = blockIdx.x;
  const int qb = idx % nqb;  idx /= nqb;
  const int h = idx % NHH;   idx /= NHH;
  const int b = idx;
  const int q0 = qb * 16;

  const bf16_t* Qh = Q + (size_t)(b * NHH + h) * LL * HDD;
  const bf16_t* Kh = K + (size_t)(b * NHH + h) * LL * HDD;
  const bf16_t* Vh = Vt + (size_t)(b * NHH + h) * HDD * LL;

  v16bf qa[4];
#pragma unroll
  for (int c = 0; c < 4; ++c)
    qa[c] = a_frag_bf16(Qh + (size_t)(q0 + col) * HDD, c * 32, lh);

  v8f o[8];
  float rmax[8], rsum[8];
#pragma unroll
  for (int t = 0; t < 8; ++t) o[t] = vzero();
#pragma unroll
  for (int r = 0; r < 8; ++r) { rmax[r] = -1e30f; rsum[r] = 0.0f; }

  const float scale = 0.08838834764831845f;  // 1/sqrt(128)
  const int kbmax = (q0 + 15) / 32;

#pragma unroll 1
  for (int kb = 0; kb <= kbmax; ++kb) {
    const int k0 = kb * 32;
    v8f s0 = vzero(), s1 = vzero();
#pragma unroll
    for (int c = 0; c < 4; ++c) {
      v16bf b0 = b_frag_bf16(Kh + (size_t)(k0 + col) * HDD, c * 32, lh);
      v16bf b1 = b_frag_bf16(Kh + (size_t)(k0 + 16 + col) * HDD, c * 32, lh);
      s0 = wmma_bf16(qa[c], b0, s0);
      s1 = wmma_bf16(qa[c], b1, s1);
    }
    // online softmax: row (r + 8*lh) owned by a 16-lane group
#pragma unroll
    for (int r = 0; r < 8; ++r) {
      const int m = q0 + r + 8 * lh;
      const float v0 = (k0 + col <= m) ? s0[r] * scale : -1e30f;
      const float v1 = (k0 + 16 + col <= m) ? s1[r] * scale : -1e30f;
      float loc = fmaxf(v0, v1);
#pragma unroll
      for (int msk = 1; msk < 16; msk <<= 1) loc = fmaxf(loc, __shfl_xor(loc, msk, 32));
      const float nm = fmaxf(rmax[r], loc);
      const float alpha = __expf(rmax[r] - nm);
      rmax[r] = nm;
      const float p0 = __expf(v0 - nm);
      const float p1 = __expf(v1 - nm);
      float ps = p0 + p1;
#pragma unroll
      for (int msk = 1; msk < 16; msk <<= 1) ps += __shfl_xor(ps, msk, 32);
      rsum[r] = rsum[r] * alpha + ps;
#pragma unroll
      for (int t = 0; t < 8; ++t) o[t][r] *= alpha;
      const int row = r + 8 * lh;
      Pl[row][col] = (__bf16)p0;
      Pl[row][col + 16] = (__bf16)p1;
    }
    __syncthreads();
    // P as A-fragment (rows-per-lane) out of LDS
    v8bf lo = *(const v8bf*)&Pl[col][lh * 8];
    v8bf hi = *(const v8bf*)&Pl[col][16 + lh * 8];
    v16bf pa;
#pragma unroll
    for (int i = 0; i < 8; ++i) { pa[i] = lo[i]; pa[i + 8] = hi[i]; }
    __syncthreads();
#pragma unroll
    for (int t = 0; t < 8; ++t) {
      v16bf vb = b_frag_bf16(Vh + (size_t)(t * 16 + col) * LL, k0, lh);
      o[t] = wmma_bf16(pa, vb, o[t]);
    }
  }

#pragma unroll
  for (int t = 0; t < 8; ++t) {
    const int d = t * 16 + col;
#pragma unroll
    for (int r = 0; r < 8; ++r) {
      const int l = q0 + r + 8 * lh;
      const float res = o[t][r] / rsum[r];
      Y[((size_t)(b * LL + l)) * CC + h * HDD + d] = (__bf16)res;
    }
  }
}

extern "C" void kernel_launch(void* const* d_in, const int* in_sizes, int n_in,
                              void* d_out, int out_size, void* d_ws, size_t ws_size,
                              hipStream_t stream) {
  (void)in_sizes; (void)n_in; (void)out_size; (void)ws_size;
  // setup_inputs order: x, rope, mask, max_seq_length, wq, wk, wv, wo
  const float* x = (const float*)d_in[0];
  const float* rope = (const float*)d_in[1];
  const float* wq = (const float*)d_in[4];
  const float* wk = (const float*)d_in[5];
  const float* wv = (const float*)d_in[6];
  const float* wo = (const float*)d_in[7];
  float* out = (float*)d_out;

  char* ws = (char*)d_ws;
  const size_t seg = (size_t)BB * LL * CC * sizeof(bf16_t);  // 32 MB
  bf16_t* Qb = (bf16_t*)(ws);
  bf16_t* Kb = (bf16_t*)(ws + seg);
  bf16_t* Vt = (bf16_t*)(ws + 2 * seg);
  bf16_t* Yb = (bf16_t*)(ws + 3 * seg);

  dim3 gg(MM / 128, NN / 128);
  dim3 blk(256);
  hipLaunchKernelGGL(HIP_KERNEL_NAME(gemm_wmma<0>), gg, blk, 0, stream,
                     (const void*)x, wq, (void*)Qb, rope);
  hipLaunchKernelGGL(HIP_KERNEL_NAME(gemm_wmma<0>), gg, blk, 0, stream,
                     (const void*)x, wk, (void*)Kb, rope);
  hipLaunchKernelGGL(HIP_KERNEL_NAME(gemm_wmma<2>), gg, blk, 0, stream,
                     (const void*)x, wv, (void*)Vt, (const float*)nullptr);
  hipLaunchKernelGGL(HIP_KERNEL_NAME(flash_attn_wmma), dim3(BB * NHH * (LL / 16)),
                     dim3(32), 0, stream, (const bf16_t*)Qb, (const bf16_t*)Kb,
                     (const bf16_t*)Vt, Yb);
  hipLaunchKernelGGL(HIP_KERNEL_NAME(gemm_wmma<3>), gg, blk, 0, stream,
                     (const void*)Yb, wo, (void*)out, (const float*)nullptr);
}
